// PromptPool_4647154614542
// MI455X (gfx1250) — compile-verified, matrix operand order checked by reading, not run on hardware
//
#include <hip/hip_runtime.h>
#include <hip/hip_bf16.h>
#include <stdint.h>

// Problem constants (from reference setup_inputs)
#define B_ 4096
#define P_ 1024
#define L_ 8
#define D_ 768
#define K_ 5

typedef __bf16 v16bf __attribute__((ext_vector_type(16)));
typedef __bf16 v8bf  __attribute__((ext_vector_type(8)));
typedef float  v8f   __attribute__((ext_vector_type(8)));

// ---------- float -> bf16, round-to-nearest-even (bit-exact, no __bf16 math) ----
__device__ __forceinline__ unsigned short f2bf_rne(float f) {
  union { float f; unsigned int u; } cv; cv.f = f;
  unsigned int u = cv.u;
  u += 0x7fffu + ((u >> 16) & 1u);
  return (unsigned short)(u >> 16);
}

// ---------- L2-normalize each row, emit bf16 (one 256-thread block per row) ----
__global__ void pp_normalize_bf16(const float* __restrict__ in,
                                  unsigned short* __restrict__ out,
                                  int D) {
  __shared__ float red[256];
  const int row = blockIdx.x;
  const int tid = threadIdx.x;
  const float* r = in + (size_t)row * D;
  float ss = 0.f;
  for (int i = tid; i < D; i += 256) { float x = r[i]; ss += x * x; }
  red[tid] = ss;
  __syncthreads();
  for (int s = 128; s > 0; s >>= 1) {
    if (tid < s) red[tid] += red[tid + s];
    __syncthreads();
  }
  const float inv = 1.0f / fmaxf(sqrtf(red[0]), 1e-12f);
  unsigned short* o = out + (size_t)row * D;
  for (int i = tid; i < D; i += 256) o[i] = f2bf_rne(r[i] * inv);
}

// ---------- similarity GEMM: one wave per 16x16 tile, v_wmma_f32_16x16x32_bf16 --
// A (q) 16x32 layout: lane = M + 16*half; half=0 -> K {0..7,16..23}, half=1 -> {8..15,24..31}
// B (k^T) 32x16: symmetric, lane = N + 16*half over the same K sub-blocks.
__global__ void pp_sim_gemm_wmma(const unsigned short* __restrict__ qn,
                                 const unsigned short* __restrict__ kn,
                                 float* __restrict__ sim) {
  const int TN   = P_ / 16;                 // 64 tiles along N
  const int wave = threadIdx.x >> 5;        // 4 waves / block
  const int lane = threadIdx.x & 31;
  const int t    = blockIdx.x * 4 + wave;   // tile id
  const int tm   = t / TN;
  const int tn   = t % TN;
  const int half = lane >> 4;
  const int r    = lane & 15;

  const unsigned short* arow = qn + (size_t)(tm * 16 + r) * D_;
  const unsigned short* brow = kn + (size_t)(tn * 16 + r) * D_;

  v8f c = {};
  #pragma unroll 4
  for (int kk = 0; kk < D_; kk += 32) {
    const int k0 = kk + half * 8;
    v8bf alo = *(const v8bf*)(arow + k0);
    v8bf ahi = *(const v8bf*)(arow + k0 + 16);
    v8bf blo = *(const v8bf*)(brow + k0);
    v8bf bhi = *(const v8bf*)(brow + k0 + 16);
    v16bf a, b;
    #pragma unroll
    for (int i = 0; i < 8; i++) {
      a[i] = alo[i]; a[i + 8] = ahi[i];
      b[i] = blo[i]; b[i + 8] = bhi[i];
    }
    // 8 args: (neg_a, A, neg_b, B, c_mod, C, reuse_a, reuse_b)
    c = __builtin_amdgcn_wmma_f32_16x16x32_bf16(false, a, false, b,
                                                (short)0, c, false, false);
  }

  // C/D layout: lane 0-15 -> (M=v, N=lane); lane 16-31 -> (M=v+8, N=lane-16)
  float* op = sim + (size_t)(tm * 16) * P_ + (size_t)tn * 16 + r;
  #pragma unroll
  for (int v = 0; v < 8; v++) op[(size_t)(v + 8 * half) * P_] = c[v];
}

// ---------- stable top-5 (descending, earlier index wins ties) per row --------
__global__ void pp_topk5(const float* __restrict__ sim,
                         float* __restrict__ tks,
                         int* __restrict__ tki) {
  const int b = blockIdx.x * blockDim.x + threadIdx.x;
  if (b >= B_) return;
  const float* row = sim + (size_t)b * P_;
  float v0 = -3e38f, v1 = -3e38f, v2 = -3e38f, v3 = -3e38f, v4 = -3e38f;
  int   i0 = 0, i1 = 0, i2 = 0, i3 = 0, i4 = 0;
  for (int j = 0; j < P_; j++) {
    const float s = row[j];
    if (s > v4) {                                    // strict > keeps ties stable
      if (s > v0)      { v4=v3;i4=i3; v3=v2;i3=i2; v2=v1;i2=i1; v1=v0;i1=i0; v0=s;i0=j; }
      else if (s > v1) { v4=v3;i4=i3; v3=v2;i3=i2; v2=v1;i2=i1; v1=s;i1=j; }
      else if (s > v2) { v4=v3;i4=i3; v3=v2;i3=i2; v2=s;i2=j; }
      else if (s > v3) { v4=v3;i4=i3; v3=s;i3=j; }
      else             { v4=s;i4=j; }
    }
  }
  float* o  = tks + (size_t)b * K_;
  int*   oi = tki + (size_t)b * K_;
  o[0]=v0; o[1]=v1; o[2]=v2; o[3]=v3; o[4]=v4;
  oi[0]=i0; oi[1]=i1; oi[2]=i2; oi[3]=i3; oi[4]=i4;
}

// ---------- gather prompt_pool[idx] -> selected (one block per (b,k) tile) ----
// Tile = 8*768 floats = 1536 float4; 256 threads x 6 float4 each.
__global__ void pp_gather(const float* __restrict__ pool,
                          const int* __restrict__ idx,
                          float* __restrict__ sel) {
  const int b = blockIdx.x / K_;
  const int k = blockIdx.x % K_;
  const int p = idx[(size_t)b * K_ + k];
  const float4* src = (const float4*)(pool + (size_t)p * (L_ * D_));
  float4* dst = (float4*)(sel + ((size_t)b * K_ + k) * (L_ * D_));
  #pragma unroll
  for (int i = 0; i < 6; i++)
    dst[threadIdx.x + i * 256] = src[threadIdx.x + i * 256];
}

extern "C" void kernel_launch(void* const* d_in, const int* in_sizes, int n_in,
                              void* d_out, int out_size, void* d_ws, size_t ws_size,
                              hipStream_t stream) {
  (void)in_sizes; (void)n_in; (void)out_size; (void)ws_size;

  const float* query = (const float*)d_in[0];   // [B, D]
  const float* pkey  = (const float*)d_in[1];   // [P, D]
  const float* ppool = (const float*)d_in[2];   // [P, L, D]

  float* out = (float*)d_out;
  float* sel = out;                                        // [B, K*L, D]
  float* sim = out + (size_t)B_ * K_ * L_ * D_;            // [B, P]
  float* tks = sim + (size_t)B_ * P_;                      // [B, K]

  char* ws = (char*)d_ws;
  unsigned short* qn = (unsigned short*)ws;                            // bf16 [B, D]
  unsigned short* kn = (unsigned short*)(ws + (size_t)B_ * D_ * 2);    // bf16 [P, D]
  int* tki = (int*)(ws + (size_t)B_ * D_ * 2 + (size_t)P_ * D_ * 2);   // [B, K]

  pp_normalize_bf16<<<B_, 256, 0, stream>>>(query, qn, D_);
  pp_normalize_bf16<<<P_, 256, 0, stream>>>(pkey, kn, D_);

  const int tiles = (B_ / 16) * (P_ / 16);                 // 16384 tiles
  pp_sim_gemm_wmma<<<tiles / 4, 128, 0, stream>>>(qn, kn, sim);

  pp_topk5<<<B_ / 256, 256, 0, stream>>>(sim, tks, tki);

  pp_gather<<<B_ * K_, 256, 0, stream>>>(ppool, tki, sel);
}
